// LearnedBeamformer_21191368639232
// MI455X (gfx1250) — matrix operand order repaired and checked by reading, compiled, not run on hardware
//
#include <hip/hip_runtime.h>

typedef __attribute__((ext_vector_type(16))) _Float16 v16h;
typedef __attribute__((ext_vector_type(8)))  float    v8f;

#define NMICS  5
#define NFREQ  257
#define TLEN   512
#define BB     8
#define NBIRD  8
#define BN     64          // BB*NBIRD
#define C1     128
#define C2     256
#define K1REAL 771         // 257*3
#define K1CH   25          // ceil(771/32)
#define K2REAL 384         // 128*3
#define K2CH   12          // 384/32
#define TW     4           // t-tiles (N=16 each) per conv block

__constant__ float MICX[5] = {0.f, 1000.f, 0.f, 1000.f, 500.f};
__constant__ float MICY[5] = {0.f, 0.f, 1000.f, 1000.f, 500.f};
__constant__ float MICZ[5] = {500.f, 500.f, 500.f, 500.f, 0.f};

// ---------------------------------------------------------------------------
// Stage 1: steering-vector MLP. One thread per (b, bird, freq).
// ---------------------------------------------------------------------------
__global__ void steering_kernel(const float* __restrict__ pos,
                                const float* __restrict__ w1, const float* __restrict__ b1,
                                const float* __restrict__ w2, const float* __restrict__ b2,
                                float* __restrict__ wrwi) {
  __shared__ float sw1[640], sw2[640], sb1[64], sb2[10];
  int tid = threadIdx.x;
  for (int i = tid; i < 640; i += 256) { sw1[i] = w1[i]; sw2[i] = w2[i]; }
  if (tid < 64) sb1[tid] = b1[tid];
  if (tid < 10) sb2[tid] = b2[tid];
  __syncthreads();

  int idx = blockIdx.x * 256 + tid;           // bn*257 + f
  if (idx >= BN * NFREQ) return;
  int f  = idx % NFREQ;
  int bn = idx / NFREQ;

  float px = pos[bn*3+0] * 1000.f;
  float py = pos[bn*3+1] * 1000.f;
  float pz = pos[bn*3+2] * 500.f;
  float d[5];
#pragma unroll
  for (int m = 0; m < 5; ++m) {
    float dx = px - MICX[m], dy = py - MICY[m], dz = pz - MICZ[m];
    d[m] = sqrtf(dx*dx + dy*dy + dz*dz);
  }
  float freq = (float)f * (24414.0f * 0.5f / 256.0f);
  float feat[10];
#pragma unroll
  for (int m = 0; m < 5; ++m) {
    float delay = (d[m] - d[0]) * (1.0f / 343000.0f);
    float ph = -2.0f * 3.14159265358979f * freq * delay;
    feat[m]   = __cosf(ph);
    feat[m+5] = __sinf(ph);
  }
  float r[10];
#pragma unroll
  for (int k = 0; k < 10; ++k) r[k] = sb2[k];
  for (int j = 0; j < 64; ++j) {
    float a = sb1[j];
#pragma unroll
    for (int i = 0; i < 10; ++i) a += feat[i] * sw1[i*64 + j];
    a = fmaxf(a, 0.f);
#pragma unroll
    for (int k = 0; k < 10; ++k) r[k] += a * sw2[j*10 + k];
  }
#pragma unroll
  for (int k = 0; k < 10; ++k) wrwi[idx*10 + k] = r[k];
}

// ---------------------------------------------------------------------------
// Stage 2: complex beamform + magnitude -> f16 [64][257][512]
// ---------------------------------------------------------------------------
__global__ void beamform_kernel(const float* __restrict__ sr, const float* __restrict__ si,
                                const float* __restrict__ wrwi, _Float16* __restrict__ mag) {
  __shared__ float w[80];                     // 8 birds x (wr[5], wi[5])
  int tid = threadIdx.x;
  int f = blockIdx.y, b = blockIdx.z;
  if (tid < 80)
    w[tid] = wrwi[((b*NBIRD + tid/10)*NFREQ + f)*10 + (tid % 10)];
  __syncthreads();

  int t = blockIdx.x * 256 + tid;
  float xr[5], xi[5];
#pragma unroll
  for (int m = 0; m < 5; ++m) {
    int off = ((b*NMICS + m)*NFREQ + f)*TLEN + t;
    xr[m] = sr[off];
    xi[m] = si[off];
  }
#pragma unroll
  for (int n = 0; n < NBIRD; ++n) {
    float br = 0.f, bi = 0.f;
#pragma unroll
    for (int m = 0; m < 5; ++m) {
      float a = w[n*10 + m], c = w[n*10 + 5 + m];
      br += xr[m]*a - xi[m]*c;
      bi += xr[m]*c + xi[m]*a;
    }
    mag[((b*NBIRD + n)*NFREQ + f)*TLEN + t] = (_Float16)sqrtf(br*br + bi*bi);
  }
}

// ---------------------------------------------------------------------------
// Pack conv weights (OIH f32) into per-lane WMMA A-operand tiles, f16, in this code's k order.
// Packed A layout (16x32 MxK, wave32): lane = (half<<4)|m;  element e:
//   K = half*16 + e  (contiguous, not AMD's table order; B reads match)
// ---------------------------------------------------------------------------
__global__ void pack_weights_kernel(const float* __restrict__ w, _Float16* __restrict__ out,
                                    int n_kc, int Cin, int Kreal, int total) {
  int idx = blockIdx.x * 256 + threadIdx.x;
  if (idx >= total) return;
  int e    = idx & 15;
  int lane = (idx >> 4) & 31;
  int rest = idx >> 9;
  int kc   = rest % n_kc;
  int mt   = rest / n_kc;
  int hl = lane >> 4;
  int j  = e >> 1;
  int kk = hl*16 + 2*j + (e & 1);
  int r = kc*32 + kk;
  float v = 0.f;
  if (r < Kreal) {
    int c = mt*16 + (lane & 15);
    int fq = r / 3, k = r - 3*fq;
    v = w[(c*Cin + fq)*3 + k];
  }
  out[idx] = (_Float16)v;
}

// ---------------------------------------------------------------------------
// conv1 implicit GEMM, TW=4 t-tiles per block (N=64), 8 waves = 8 M-tiles.
// Staging: thread owns one K-row (K=tid>>3) and an 8-wide t segment of one
// t-tile; (f,k)=divmod(r,3) kept incrementally across K-chunks (r += 32 =>
// f += 10+(k>=1), k = (k==0)?2:k-1). Bounds handled branchlessly via
// clamp+select. B tile stored in B-matrix layout [n][K] -> lane reads 32
// contiguous bytes (2x ds_load_b128). A fragment: 1x 32B contiguous global
// load per lane, reused across TW WMMAs.
// ---------------------------------------------------------------------------
__global__ void __launch_bounds__(256)
conv1_wmma_kernel(const _Float16* __restrict__ X, const _Float16* __restrict__ A,
                  const float* __restrict__ bias, float* __restrict__ Y) {
  __shared__ alignas(32) _Float16 Bs[TW][16][48];
  int tid  = threadIdx.x;
  int lane = tid & 31, wave = tid >> 5;
  int tbase = blockIdx.x * (16*TW), bn = blockIdx.y;
  int hl = lane >> 4, n = lane & 15;

  // staging assignment
  int sK  = tid >> 3;               // 0..31 : K row within chunk
  int stt = (tid >> 1) & 3;         // t-tile
  int snb = (tid & 1) * 8;          // n sub-range base
  int sfq = sK / 3;                 // f for r = sK (updated incrementally)
  int sk  = sK - 3*sfq;             // tap k

  v8f acc[TW] = {};

  for (int kc = 0; kc < K1CH; ++kc) {
    // ---- stage 4 B tiles (2048 halves), branchless ----
    {
      int r = kc*32 + sK;
      int fqc = (sfq > 256) ? 256 : sfq;
      const _Float16* src = X + (bn*NFREQ + fqc)*TLEN;
      int t = tbase + stt*16 + snb + sk - 1;
      int rvalid = (r < K1REAL);
#pragma unroll
      for (int q = 0; q < 8; ++q) {
        int tq = t + q;
        int ok = rvalid & ((unsigned)tq < (unsigned)TLEN);
        _Float16 x = src[ok ? tq : 0];
        Bs[stt][snb + q][sK] = ok ? x : (_Float16)0.f;
      }
      sfq += 10 + (sk >= 1);
      sk = (sk == 0) ? 2 : (sk - 1);
    }
    __syncthreads();
    v16h a = *(const v16h*)(A + ((wave*K1CH + kc)*32 + lane)*16);
#pragma unroll
    for (int tt = 0; tt < TW; ++tt) {
      v16h bm = *(const v16h*)(&Bs[tt][n][hl*16]);
      acc[tt] = __builtin_amdgcn_wmma_f32_16x16x32_f16(false, a, false, bm,
                                                       (short)0, acc[tt], false, false);
    }
    __syncthreads();
  }

#pragma unroll
  for (int v = 0; v < 8; ++v) {
    int c = wave*16 + hl*8 + v;
    float bv = bias[c];
#pragma unroll
    for (int tt = 0; tt < TW; ++tt)
      Y[(bn*C1 + c)*TLEN + tbase + tt*16 + n] = acc[tt][v] + bv;
  }
}

// ---------------------------------------------------------------------------
// conv2: 16 M-tiles; each wave owns 2 M-tiles x TW t-tiles (8 accumulators).
// K = 384 exactly (no r-bound needed).
// ---------------------------------------------------------------------------
__global__ void __launch_bounds__(256)
conv2_wmma_kernel(const _Float16* __restrict__ X, const _Float16* __restrict__ A,
                  const float* __restrict__ bias, float* __restrict__ Y) {
  __shared__ alignas(32) _Float16 Bs[TW][16][48];
  int tid  = threadIdx.x;
  int lane = tid & 31, wave = tid >> 5;
  int tbase = blockIdx.x * (16*TW), bn = blockIdx.y;
  int hl = lane >> 4, n = lane & 15;
  int mt0 = wave*2, mt1 = wave*2 + 1;

  int sK  = tid >> 3;
  int stt = (tid >> 1) & 3;
  int snb = (tid & 1) * 8;
  int sfq = sK / 3;
  int sk  = sK - 3*sfq;

  v8f acc0[TW] = {}, acc1[TW] = {};

  for (int kc = 0; kc < K2CH; ++kc) {
    {
      const _Float16* src = X + (bn*C1 + sfq)*TLEN;
      int t = tbase + stt*16 + snb + sk - 1;
#pragma unroll
      for (int q = 0; q < 8; ++q) {
        int tq = t + q;
        int ok = ((unsigned)tq < (unsigned)TLEN);
        _Float16 x = src[ok ? tq : 0];
        Bs[stt][snb + q][sK] = ok ? x : (_Float16)0.f;
      }
      sfq += 10 + (sk >= 1);
      sk = (sk == 0) ? 2 : (sk - 1);
    }
    __syncthreads();
    v16h a0 = *(const v16h*)(A + ((mt0*K2CH + kc)*32 + lane)*16);
    v16h a1 = *(const v16h*)(A + ((mt1*K2CH + kc)*32 + lane)*16);
#pragma unroll
    for (int tt = 0; tt < TW; ++tt) {
      v16h bm = *(const v16h*)(&Bs[tt][n][hl*16]);
      acc0[tt] = __builtin_amdgcn_wmma_f32_16x16x32_f16(false, a0, false, bm,
                                                        (short)0, acc0[tt], false, false);
      acc1[tt] = __builtin_amdgcn_wmma_f32_16x16x32_f16(false, a1, false, bm,
                                                        (short)0, acc1[tt], false, false);
    }
    __syncthreads();
  }

#pragma unroll
  for (int v = 0; v < 8; ++v) {
    int c0 = mt0*16 + hl*8 + v;
    int c1 = mt1*16 + hl*8 + v;
    float b0 = bias[c0], b1 = bias[c1];
#pragma unroll
    for (int tt = 0; tt < TW; ++tt) {
      Y[(bn*C2 + c0)*TLEN + tbase + tt*16 + n] = acc0[tt][v] + b0;
      Y[(bn*C2 + c1)*TLEN + tbase + tt*16 + n] = acc1[tt][v] + b1;
    }
  }
}

// ---------------------------------------------------------------------------
// BatchNorm statistics over (64 samples x 512 t) per channel -> scale/shift
// ---------------------------------------------------------------------------
__global__ void bn_stats_kernel(const float* __restrict__ y, const float* __restrict__ g,
                                const float* __restrict__ be, float* __restrict__ stats, int C) {
  __shared__ float s1[256], s2[256];
  int c = blockIdx.x, tid = threadIdx.x;
  float a = 0.f, b = 0.f;
  for (int i = tid; i < BN*TLEN; i += 256) {
    int bnI = i >> 9, t = i & 511;
    float v = y[(bnI*C + c)*TLEN + t];
    a += v; b += v*v;
  }
  s1[tid] = a; s2[tid] = b;
  __syncthreads();
  for (int s = 128; s > 0; s >>= 1) {
    if (tid < s) { s1[tid] += s1[tid+s]; s2[tid] += s2[tid+s]; }
    __syncthreads();
  }
  if (tid == 0) {
    float m   = s1[0] * (1.0f/32768.0f);
    float var = s2[0] * (1.0f/32768.0f) - m*m;
    float sc  = g[c] * rsqrtf(var + 1e-5f);
    stats[2*c]   = sc;
    stats[2*c+1] = be[c] - m*sc;
  }
}

// normalize + relu -> f16 (feeds conv2)
__global__ void bn_relu_kernel(const float* __restrict__ y, const float* __restrict__ stats,
                               _Float16* __restrict__ x, int C, int total) {
  int i = blockIdx.x * 256 + threadIdx.x;
  if (i >= total) return;
  int c = (i >> 9) % C;
  float v = y[i]*stats[2*c] + stats[2*c+1];
  x[i] = (_Float16)fmaxf(v, 0.f);
}

// normalize + relu + mean over t -> output embedding (one wave per (bn,c))
__global__ void final_mean_kernel(const float* __restrict__ y, const float* __restrict__ stats,
                                  float* __restrict__ out) {
  int bnc = blockIdx.x;                        // bn*256 + c
  int c = bnc & 255;
  int lane = threadIdx.x;
  float sc = stats[2*c], sh = stats[2*c+1];
  float s = 0.f;
  for (int t = lane; t < TLEN; t += 32)
    s += fmaxf(y[bnc*TLEN + t]*sc + sh, 0.f);
#pragma unroll
  for (int off = 16; off > 0; off >>= 1) s += __shfl_xor(s, off, 32);
  if (lane == 0) out[bnc] = s * (1.0f/512.0f);
}

// ---------------------------------------------------------------------------
// Workspace layout (bytes), 256-aligned
// ---------------------------------------------------------------------------
static constexpr size_t OFF_WRWI = 0;                                  // 64*257*10 f32
static constexpr size_t OFF_MAG  = 658176;                             // 64*257*512 f16
static constexpr size_t OFF_A1   = OFF_MAG + 16842752;                 // 8*25*512  f16
static constexpr size_t OFF_A2   = OFF_A1  + 204800;                   // 16*12*512 f16
static constexpr size_t OFF_Y1   = OFF_A2  + 196608;                   // 64*128*512 f32
static constexpr size_t OFF_X2   = OFF_Y1  + 16777216;                 // 64*128*512 f16
static constexpr size_t OFF_Y2   = OFF_X2  + 8388608;                  // 64*256*512 f32
static constexpr size_t OFF_ST1  = OFF_Y2  + 33554432;                 // 128*2 f32
static constexpr size_t OFF_ST2  = OFF_ST1 + 1024;                     // 256*2 f32

extern "C" void kernel_launch(void* const* d_in, const int* in_sizes, int n_in,
                              void* d_out, int out_size, void* d_ws, size_t ws_size,
                              hipStream_t stream) {
  const float* stft_real = (const float*)d_in[0];
  const float* stft_imag = (const float*)d_in[1];
  const float* pos       = (const float*)d_in[2];
  const float* w1  = (const float*)d_in[3];
  const float* b1  = (const float*)d_in[4];
  const float* w2  = (const float*)d_in[5];
  const float* b2  = (const float*)d_in[6];
  const float* cw1 = (const float*)d_in[7];
  const float* cb1 = (const float*)d_in[8];
  const float* g1  = (const float*)d_in[9];
  const float* be1 = (const float*)d_in[10];
  const float* cw2 = (const float*)d_in[11];
  const float* cb2 = (const float*)d_in[12];
  const float* g2  = (const float*)d_in[13];
  const float* be2 = (const float*)d_in[14];

  char* ws = (char*)d_ws;
  float*    wrwi = (float*)(ws + OFF_WRWI);
  _Float16* mag  = (_Float16*)(ws + OFF_MAG);
  _Float16* A1   = (_Float16*)(ws + OFF_A1);
  _Float16* A2   = (_Float16*)(ws + OFF_A2);
  float*    y1   = (float*)(ws + OFF_Y1);
  _Float16* x2   = (_Float16*)(ws + OFF_X2);
  float*    y2   = (float*)(ws + OFF_Y2);
  float*    st1  = (float*)(ws + OFF_ST1);
  float*    st2  = (float*)(ws + OFF_ST2);

  // 1) steering weights + WMMA weight packing
  steering_kernel<<<(BN*NFREQ + 255)/256, 256, 0, stream>>>(pos, w1, b1, w2, b2, wrwi);
  pack_weights_kernel<<<(8*K1CH*512 + 255)/256, 256, 0, stream>>>(cw1, A1, K1CH, NFREQ, K1REAL, 8*K1CH*512);
  pack_weights_kernel<<<(16*K2CH*512 + 255)/256, 256, 0, stream>>>(cw2, A2, K2CH, C1, K2REAL, 16*K2CH*512);
  // 2) beamform + magnitude
  beamform_kernel<<<dim3(TLEN/256, NFREQ, BB), 256, 0, stream>>>(stft_real, stft_imag, wrwi, mag);
  // 3) conv1 (WMMA) + BN + relu
  conv1_wmma_kernel<<<dim3(TLEN/(16*TW), BN), 256, 0, stream>>>(mag, A1, cb1, y1);
  bn_stats_kernel<<<C1, 256, 0, stream>>>(y1, g1, be1, st1, C1);
  bn_relu_kernel<<<(BN*C1*TLEN)/256, 256, 0, stream>>>(y1, st1, x2, C1, BN*C1*TLEN);
  // 4) conv2 (WMMA) + BN + relu + mean
  conv2_wmma_kernel<<<dim3(TLEN/(16*TW), BN), 256, 0, stream>>>(x2, A2, cb2, y2);
  bn_stats_kernel<<<C2, 256, 0, stream>>>(y2, g2, be2, st2, C2);
  final_mean_kernel<<<BN*C2, 32, 0, stream>>>(y2, st2, (float*)d_out);
}
